// MambaMixer_42623255446107
// MI455X (gfx1250) — matrix operand order repaired and checked by reading, not compiled, not run on hardware
//
#include <hip/hip_runtime.h>
#include <hip/hip_bf16.h>
#include <math.h>

// MambaMixer (bidirectional) for MI455X / gfx1250.
// wave32 + v_wmma_f32_16x16x32_bf16 (hi/lo split GEMMs) + TDM tensor_load_to_lds
// double-buffered tile feed.  Workspace requirement: ~78 MiB.

typedef __attribute__((ext_vector_type(16))) __bf16 v16bf;
typedef __attribute__((ext_vector_type(8)))  __bf16 v8bf;
typedef __attribute__((ext_vector_type(8)))  float  v8f;
typedef __attribute__((ext_vector_type(4)))  float  f4;
typedef __attribute__((ext_vector_type(4)))  unsigned int u32x4;
typedef __attribute__((ext_vector_type(8)))  int    i32x8;
typedef __attribute__((ext_vector_type(4)))  int    i32x4;

#if __has_builtin(__builtin_amdgcn_tensor_load_to_lds) && __has_builtin(__builtin_amdgcn_s_wait_tensorcnt)
#define HAVE_TDM 1
#else
#define HAVE_TDM 0
#endif

#define ROWB 80                      // LDS row stride: 64B data + 16B pad (bank spread, 16B aligned)
#define A_TILE_B (128 * ROWB)        // 10240 B
#define B_TILE_B (64 * ROWB)         //  5120 B
#define BUF_B (2 * A_TILE_B + 2 * B_TILE_B)   // Ah | Al | Bh | Bl

__device__ __forceinline__ float silu_f(float x) {
    return x / (1.0f + __expf(-x));
}

#if HAVE_TDM
// Issue one 2D TDM tile load: rows x 32 bf16 elements, row stride ld (elems).
// rows_rem < rows makes trailing rows read as zero (free N-edge handling).
// LDS padding: +4 DWORDs every 16 DWORDs -> 80B LDS row stride.
__device__ __forceinline__ void tdm_tile_load(unsigned lds_addr, const __bf16* g,
                                              int rows, int rows_rem, int ld)
{
    unsigned long long ga = (unsigned long long)(uintptr_t)g;
    u32x4 g0;
    g0[0] = 1u;                                            // count=1 (valid), user mode
    g0[1] = lds_addr;                                      // LDS byte address
    g0[2] = (unsigned)ga;                                  // global_addr[31:0]
    g0[3] = (unsigned)((ga >> 32) & 0x01ffffffu) | (2u << 30);  // global_addr[56:32] | type=2
    i32x8 g1;
    unsigned td1 = (unsigned)rows_rem;
    g1[0] = (int)((1u << 16) | (1u << 20) | (3u << 22) | (3u << 25));
    //         data_size=2B | pad_enable | pad_interval=16dw | pad_amount=4dw
    g1[1] = (int)(32u << 16);                              // tensor_dim0[15:0]=32
    g1[2] = (int)((td1 & 0xffffu) << 16);                  // tensor_dim0 hi=0 | tensor_dim1 lo
    g1[3] = (int)((td1 >> 16) | (32u << 16));              // tensor_dim1 hi | tile_dim0=32
    g1[4] = rows;                                          // tile_dim1 | tile_dim2=0
    g1[5] = ld;                                            // tensor_dim0_stride[31:0]
    g1[6] = 0;                                             // stride hi | dim1_stride lo
    g1[7] = 0;
    i32x4 gz4 = {0, 0, 0, 0};
    i32x8 gz8 = {0, 0, 0, 0, 0, 0, 0, 0};
    __builtin_amdgcn_tensor_load_to_lds(g0, g1, gz4, gz4, gz8, 0);
}
#endif

// ---------------------------------------------------------------------------
// GEMM: C[M,N] = (Ah+Al)[M,K] * (Bh+Bl)[N,K]^T, planes are bf16, C fp32.
// 3 WMMAs per 16x16x32 step (Ah*Bh + Ah*Bl + Al*Bh).  Block 256 thr = 8 waves,
// block tile 128x64, wave tile 32x32 (2x2 accumulators), K-step 32.
// TDM double-buffered tile feed (wave 0 issues; s_wait_tensorcnt + barrier).
// ---------------------------------------------------------------------------
__global__ void __launch_bounds__(256)
gemm_planes_kernel(const __bf16* __restrict__ Ah, const __bf16* __restrict__ Al, int lda,
                   const __bf16* __restrict__ Bh, const __bf16* __restrict__ Bl, int ldb,
                   float* __restrict__ C, int ldc, int N, int K)
{
    __shared__ __align__(16) char lds[2 * BUF_B];

    const int tid  = threadIdx.x;
    const int lane = tid & 31;
    const int wid  = tid >> 5;
    const int wm   = wid & 3;
    const int wn   = wid >> 2;
    const int m0   = blockIdx.y * 128;
    const int n0   = blockIdx.x * 64;
    const int lr   = lane & 15;
    const int half = lane >> 4;

    int remB = N - n0; if (remB > 64) remB = 64;

    v8f acc[2][2] = {};

#if HAVE_TDM
    const unsigned lbase = (unsigned)(uintptr_t)&lds[0];
#endif

    auto issue = [&](int k0, int buf) {
#if HAVE_TDM
        if (wid == 0) {
            unsigned o = lbase + (unsigned)(buf * BUF_B);
            tdm_tile_load(o,                          Ah + (size_t)m0 * lda + k0, 128, 128, lda);
            tdm_tile_load(o + A_TILE_B,               Al + (size_t)m0 * lda + k0, 128, 128, lda);
            tdm_tile_load(o + 2 * A_TILE_B,           Bh + (size_t)n0 * ldb + k0,  64, remB, ldb);
            tdm_tile_load(o + 2 * A_TILE_B + B_TILE_B, Bl + (size_t)n0 * ldb + k0, 64, remB, ldb);
        }
#else
        // Cooperative fallback: 16B chunks, all 256 threads.
        char* dst = lds + buf * BUF_B;
        // A tiles: 128 rows x 4 chunks, hi then lo plane (2 chunks / thread each).
        #pragma unroll
        for (int t = 0; t < 2; ++t) {
            int i = tid + t * 256;             // 0..511
            int r = i >> 2, c = i & 3;
            *(f4*)(dst + r * ROWB + c * 16) =
                *(const f4*)((const char*)(Ah + (size_t)(m0 + r) * lda + k0) + c * 16);
            *(f4*)(dst + A_TILE_B + r * ROWB + c * 16) =
                *(const f4*)((const char*)(Al + (size_t)(m0 + r) * lda + k0) + c * 16);
        }
        // B tiles: 64 rows x 4 chunks (1 chunk / thread each).
        {
            int r = tid >> 2, c = tid & 3;
            f4 vh = {0.f, 0.f, 0.f, 0.f}, vl = vh;
            if (r < remB) {
                vh = *(const f4*)((const char*)(Bh + (size_t)(n0 + r) * ldb + k0) + c * 16);
                vl = *(const f4*)((const char*)(Bl + (size_t)(n0 + r) * ldb + k0) + c * 16);
            }
            *(f4*)(dst + 2 * A_TILE_B + r * ROWB + c * 16) = vh;
            *(f4*)(dst + 2 * A_TILE_B + B_TILE_B + r * ROWB + c * 16) = vl;
        }
#endif
    };

    issue(0, 0);
#if HAVE_TDM
    __builtin_amdgcn_s_wait_tensorcnt(0);
#endif
    __syncthreads();

    int p = 0;
    for (int k0 = 0; k0 < K; k0 += 32) {
        if (k0 + 32 < K) issue(k0 + 32, p ^ 1);   // async prefetch into other buffer

        const char* bufp = lds + p * BUF_B;

        // Fragments: packed bf16 K-runs -> v16bf halves, per ISA VGPR layouts.
        union bfrag { v16bf v; v8bf h[2]; };
        bfrag fAh[2], fAl[2], fBh[2], fBl[2];
        #pragma unroll
        for (int mt = 0; mt < 2; ++mt) {
            // A: lanes<16 take K 0-7 & 16-23; lanes>=16 take K 8-15 & 24-31.
            const char* rp = bufp + (wm * 32 + mt * 16 + lr) * ROWB + half * 16;
            fAh[mt].h[0] = *(const v8bf*)(rp);
            fAh[mt].h[1] = *(const v8bf*)(rp + 32);
            const char* rq = rp + A_TILE_B;
            fAl[mt].h[0] = *(const v8bf*)(rq);
            fAl[mt].h[1] = *(const v8bf*)(rq + 32);
        }
        #pragma unroll
        for (int nt = 0; nt < 2; ++nt) {
            // B: lanes<16 take K 0-7 & 16-23; lanes>=16 take K 8-15 & 24-31.
            const char* rp = bufp + 2 * A_TILE_B + (wn * 32 + nt * 16 + lr) * ROWB + half * 16;
            fBh[nt].h[0] = *(const v8bf*)(rp);
            fBh[nt].h[1] = *(const v8bf*)(rp + 32);
            const char* rq = rp + B_TILE_B;
            fBl[nt].h[0] = *(const v8bf*)(rq);
            fBl[nt].h[1] = *(const v8bf*)(rq + 32);
        }

        #pragma unroll
        for (int mt = 0; mt < 2; ++mt) {
            #pragma unroll
            for (int nt = 0; nt < 2; ++nt) {
                acc[mt][nt] = __builtin_amdgcn_wmma_f32_16x16x32_bf16(
                    false, fAh[mt].v, false, fBh[nt].v, (short)0, acc[mt][nt], false, false);
                acc[mt][nt] = __builtin_amdgcn_wmma_f32_16x16x32_bf16(
                    false, fAh[mt].v, false, fBl[nt].v, (short)0, acc[mt][nt], false, false);
                acc[mt][nt] = __builtin_amdgcn_wmma_f32_16x16x32_bf16(
                    false, fAl[mt].v, false, fBh[nt].v, (short)0, acc[mt][nt], false, false);
            }
        }
        __syncthreads();                      // all waves done reading buf p
#if HAVE_TDM
        __builtin_amdgcn_s_wait_tensorcnt(0); // wave0 gates on prefetch completion
#endif
        __syncthreads();                      // buf p^1 visible to all
        p ^= 1;
    }

    // Store C: v8f element r -> row (lane<16 ? r : 8+r), col = lane%16.
    #pragma unroll
    for (int mt = 0; mt < 2; ++mt) {
        #pragma unroll
        for (int nt = 0; nt < 2; ++nt) {
            const int col = n0 + wn * 32 + nt * 16 + lr;
            if (col < N) {
                const int rbase = m0 + wm * 32 + mt * 16 + half * 8;
                #pragma unroll
                for (int r = 0; r < 8; ++r)
                    C[(size_t)(rbase + r) * ldc + col] = acc[mt][nt][r];
            }
        }
    }
}

// ---------------------------------------------------------------------------
// fp32 -> bf16 hi/lo plane split.  cols = 1<<colshift; src row stride srcld.
// ---------------------------------------------------------------------------
__global__ void __launch_bounds__(256)
split_bf16_kernel(const float* __restrict__ src, int srcld, int colshift,
                  __bf16* __restrict__ hi, __bf16* __restrict__ lo, int total)
{
    const int idx = blockIdx.x * 256 + threadIdx.x;
    if (idx >= total) return;
    const int r = idx >> colshift;
    const int c = idx - (r << colshift);
    const float v = src[(size_t)r * srcld + c];
    const __bf16 h = (__bf16)v;
    hi[idx] = h;
    lo[idx] = (__bf16)(v - (float)h);
}

// ---------------------------------------------------------------------------
// Depthwise causal conv (K=4) + bias + SiLU; emits fp32 + bf16 hi/lo planes.
// ---------------------------------------------------------------------------
__global__ void __launch_bounds__(256)
conv_silu_kernel(const float* __restrict__ x,
                 const float* __restrict__ w, const float* __restrict__ bias,
                 float* __restrict__ out,
                 __bf16* __restrict__ outh, __bf16* __restrict__ outl)
{
    const int idx = blockIdx.x * 256 + threadIdx.x;   // (b,l,i) flat
    const int i = idx & 2047;
    const int l = (idx >> 11) & 511;
    float acc = bias[i];
    const float* wr = w + i * 4;
    #pragma unroll
    for (int k = 0; k < 4; ++k) {
        const int ls = l - 3 + k;
        if (ls >= 0)
            acc += x[idx + (k - 3) * 2048] * wr[k];
    }
    const float s = silu_f(acc);
    out[idx] = s;
    const __bf16 h = (__bf16)s;
    outh[idx] = h;
    outl[idx] = (__bf16)(s - (float)h);
}

// ---------------------------------------------------------------------------
// Bidirectional selective scan.  Thread = (b,i,n); 16 lanes share a channel.
// Forward writes ybuf (fp32); backward adds and emits final y as bf16 planes.
// ---------------------------------------------------------------------------
__global__ void __launch_bounds__(256)
scan_kernel(const float* __restrict__ hs, const float* __restrict__ gate,
            const float* __restrict__ ssm_f, const float* __restrict__ ssm_b,
            const float* __restrict__ dpre_f, const float* __restrict__ dpre_b,
            const float* __restrict__ dtb_f, const float* __restrict__ dtb_b,
            const float* __restrict__ Alog_f, const float* __restrict__ Alog_b,
            const float* __restrict__ Dv_f, const float* __restrict__ Dv_b,
            float* __restrict__ ybuf,
            __bf16* __restrict__ yh, __bf16* __restrict__ yl)
{
    const int gid = blockIdx.x * 256 + threadIdx.x;
    const int n  = gid & 15;
    const int ch = gid >> 4;       // b*2048 + i
    const int i  = ch & 2047;
    const int b  = ch >> 11;

    const float Af = -__expf(Alog_f[i * 16 + n]);
    const float Ab = -__expf(Alog_b[i * 16 + n]);
    const float dbf = dtb_f[i], dbb = dtb_b[i];
    const float Df = Dv_f[i],  Db = Dv_b[i];

    const size_t base  = (size_t)b * 512 * 2048 + i;
    const size_t sbase = (size_t)b * 512 * 96;

    float st = 0.f;
    for (int l = 0; l < 512; ++l) {
        const size_t idx = base + (size_t)l * 2048;
        const float u  = hs[idx];
        const float dp = dpre_f[idx] + dbf;
        const float delta = (dp > 20.f) ? dp : log1pf(__expf(dp));
        const float* sr = ssm_f + sbase + (size_t)l * 96;
        st = st * __expf(delta * Af) + delta * sr[64 + n] * u;
        float v = st * sr[80 + n];
        v += __shfl_xor(v, 1, 32);
        v += __shfl_xor(v, 2, 32);
        v += __shfl_xor(v, 4, 32);
        v += __shfl_xor(v, 8, 32);
        if (n == 0)
            ybuf[idx] = (v + u * Df) * silu_f(gate[idx]);
    }

    st = 0.f;
    for (int l = 511; l >= 0; --l) {
        const size_t idx = base + (size_t)l * 2048;
        const float u  = hs[idx];
        const float dp = dpre_b[idx] + dbb;
        const float delta = (dp > 20.f) ? dp : log1pf(__expf(dp));
        const float* sr = ssm_b + sbase + (size_t)l * 96;
        st = st * __expf(delta * Ab) + delta * sr[64 + n] * u;
        float v = st * sr[80 + n];
        v += __shfl_xor(v, 1, 32);
        v += __shfl_xor(v, 2, 32);
        v += __shfl_xor(v, 4, 32);
        v += __shfl_xor(v, 8, 32);
        if (n == 0) {
            const float tot = ybuf[idx] + (v + u * Db) * silu_f(gate[idx]);
            const __bf16 h = (__bf16)tot;
            yh[idx] = h;
            yl[idx] = (__bf16)(tot - (float)h);
        }
    }
}

// ---------------------------------------------------------------------------
extern "C" void kernel_launch(void* const* d_in, const int* in_sizes, int n_in,
                              void* d_out, int out_size, void* d_ws, size_t ws_size,
                              hipStream_t stream)
{
    const float* hs0    = (const float*)d_in[0];
    const float* hs2    = (const float*)d_in[1];
    const float* inW    = (const float*)d_in[2];
    const float* inDifW = (const float*)d_in[3];
    const float* convW  = (const float*)d_in[4];
    const float* convB  = (const float*)d_in[5];
    const float* xW_f   = (const float*)d_in[6];
    const float* xW_b   = (const float*)d_in[7];
    const float* dtW_f  = (const float*)d_in[8];
    const float* dtB_f  = (const float*)d_in[9];
    const float* dtW_b  = (const float*)d_in[10];
    const float* dtB_b  = (const float*)d_in[11];
    const float* Alog_f = (const float*)d_in[12];
    const float* Alog_b = (const float*)d_in[13];
    const float* Dv_f   = (const float*)d_in[14];
    const float* Dv_b   = (const float*)d_in[15];
    const float* outW   = (const float*)d_in[16];
    float* out = (float*)d_out;

    char* W = (char*)d_ws;
    size_t o = 0;
    auto alloc = [&](size_t bytes) { char* p = W + o; o += (bytes + 255) & ~(size_t)255; return p; };

    // fp32 arena
    float* x      = (float*)alloc(8388608);   // aliased later as dpre_f
    float* gate   = (float*)alloc(8388608);
    float* hsf    = (float*)alloc(8388608);
    float* dpre_b = (float*)alloc(8388608);
    float* ybuf   = (float*)alloc(8388608);
    float* ssm_f  = (float*)alloc(393216);
    float* ssm_b  = (float*)alloc(393216);
    // bf16 planes
    __bf16* hs0h = (__bf16*)alloc(2097152);   // [hs0h..hs2l] reused as y planes
    __bf16* hs0l = (__bf16*)alloc(2097152);
    __bf16* hs2h = (__bf16*)alloc(2097152);
    __bf16* hs2l = (__bf16*)alloc(2097152);
    __bf16* inWh = (__bf16*)alloc(4194304);   // reused as hs_f planes
    __bf16* inWl = (__bf16*)alloc(4194304);
    __bf16* inDh = (__bf16*)alloc(4194304);   // reused as dt_r planes
    __bf16* inDl = (__bf16*)alloc(4194304);
    __bf16* xWfh = (__bf16*)alloc(393216);
    __bf16* xWfl = (__bf16*)alloc(393216);
    __bf16* xWbh = (__bf16*)alloc(393216);
    __bf16* xWbl = (__bf16*)alloc(393216);
    __bf16* dtWfh = (__bf16*)alloc(262144);
    __bf16* dtWfl = (__bf16*)alloc(262144);
    __bf16* dtWbh = (__bf16*)alloc(262144);
    __bf16* dtWbl = (__bf16*)alloc(262144);
    __bf16* outWh = (__bf16*)alloc(4194304);
    __bf16* outWl = (__bf16*)alloc(4194304);
    // lifetime-safe aliases
    float*  dpre_f = x;                        // x dead after conv; dpre_f written later
    __bf16* hsfh = inWh, *hsfl = inWl;         // inW planes dead after GEMM 1
    __bf16* dtrfh = inDh;                      // inDif planes dead after GEMM 2
    __bf16* dtrfl = (__bf16*)((char*)inDh + 131072);
    __bf16* dtrbh = (__bf16*)((char*)inDh + 262144);
    __bf16* dtrbl = (__bf16*)((char*)inDh + 393216);
    __bf16* yh = hs0h;                         // hs planes dead after GEMMs 1-2
    __bf16* yl = hs2h;

    const dim3 blk(256);

    // 0) split fp32 operands into bf16 hi/lo planes
    split_bf16_kernel<<<dim3(8192), blk, 0, stream>>>(inW, 1024, 10, inWh, inWl, 2097152);
    split_bf16_kernel<<<dim3(8192), blk, 0, stream>>>(inDifW + (size_t)2048 * 1024, 1024, 10, inDh, inDl, 2097152);
    split_bf16_kernel<<<dim3(4096), blk, 0, stream>>>(hs0, 1024, 10, hs0h, hs0l, 1048576);
    split_bf16_kernel<<<dim3(4096), blk, 0, stream>>>(hs2, 1024, 10, hs2h, hs2l, 1048576);
    split_bf16_kernel<<<dim3(768),  blk, 0, stream>>>(xW_f, 2048, 11, xWfh, xWfl, 196608);
    split_bf16_kernel<<<dim3(768),  blk, 0, stream>>>(xW_b, 2048, 11, xWbh, xWbl, 196608);
    split_bf16_kernel<<<dim3(512),  blk, 0, stream>>>(dtW_f, 64, 6, dtWfh, dtWfl, 131072);
    split_bf16_kernel<<<dim3(512),  blk, 0, stream>>>(dtW_b, 64, 6, dtWbh, dtWbl, 131072);
    split_bf16_kernel<<<dim3(8192), blk, 0, stream>>>(outW, 2048, 11, outWh, outWl, 2097152);

    // 1-2) input projections (1024x2048, K=1024)
    gemm_planes_kernel<<<dim3(32, 8), blk, 0, stream>>>(hs0h, hs0l, 1024, inWh, inWl, 1024, x, 2048, 2048, 1024);
    gemm_planes_kernel<<<dim3(32, 8), blk, 0, stream>>>(hs2h, hs2l, 1024, inDh, inDl, 1024, gate, 2048, 2048, 1024);

    // 3) causal conv + SiLU -> hs_f (fp32 + planes)
    conv_silu_kernel<<<dim3(8192), blk, 0, stream>>>(x, convW, convB, hsf, hsfh, hsfl);

    // 4) ssm projections (1024x96, K=2048)
    gemm_planes_kernel<<<dim3(2, 8), blk, 0, stream>>>(hsfh, hsfl, 2048, xWfh, xWfl, 2048, ssm_f, 96, 96, 2048);
    gemm_planes_kernel<<<dim3(2, 8), blk, 0, stream>>>(hsfh, hsfl, 2048, xWbh, xWbl, 2048, ssm_b, 96, 96, 2048);

    // 5) split dt_r (strided: 96-wide rows, first 64 cols)
    split_bf16_kernel<<<dim3(256), blk, 0, stream>>>(ssm_f, 96, 6, dtrfh, dtrfl, 65536);
    split_bf16_kernel<<<dim3(256), blk, 0, stream>>>(ssm_b, 96, 6, dtrbh, dtrbl, 65536);

    // 6) delta projections (1024x2048, K=64)
    gemm_planes_kernel<<<dim3(32, 8), blk, 0, stream>>>(dtrfh, dtrfl, 64, dtWfh, dtWfl, 64, dpre_f, 2048, 2048, 64);
    gemm_planes_kernel<<<dim3(32, 8), blk, 0, stream>>>(dtrbh, dtrbl, 64, dtWbh, dtWbl, 64, dpre_b, 2048, 2048, 64);

    // 7) bidirectional selective scan -> y planes
    scan_kernel<<<dim3(256), blk, 0, stream>>>(hsf, gate, ssm_f, ssm_b, dpre_f, dpre_b,
                                               dtB_f, dtB_b, Alog_f, Alog_b, Dv_f, Dv_b,
                                               ybuf, yh, yl);

    // 8) output projection (1024x1024, K=2048) -> d_out
    gemm_planes_kernel<<<dim3(16, 8), blk, 0, stream>>>(yh, yl, 2048, outWh, outWl, 2048, out, 1024, 1024, 2048);
}
